// SpatialFormer_2559800509206
// MI455X (gfx1250) — matrix operand order repaired and checked by reading, not compiled, not run on hardware
//
#include <hip/hip_runtime.h>
#include <math.h>

typedef float  v8f   __attribute__((ext_vector_type(8)));
typedef __bf16 v16bf __attribute__((ext_vector_type(16)));

#define NY 16777216ULL   // 16*256*64*64 elements of one full-size activation

__device__ __forceinline__ float gelu_exact(float x) {
  return 0.5f * x * (1.0f + erff(x * 0.70710678118654752440f));
}

// ---------------------------------------------------------------------------
// BN parameter folding: A = s/sqrt(v+eps), B = b - m*A   (bn1, fbn, gbn)
// ---------------------------------------------------------------------------
__global__ void bnprep_kernel(const float* s1, const float* b1, const float* m1, const float* v1,
                              const float* fs, const float* fb, const float* fm, const float* fv,
                              const float* gs, const float* gb, const float* gm, const float* gv,
                              float* outp) {
  int t = threadIdx.x;
  float inv = s1[t] * rsqrtf(v1[t] + 1e-5f);
  outp[t]       = inv;
  outp[256 + t] = b1[t] - m1[t] * inv;
  float finv = fs[t] * rsqrtf(fv[t] + 1e-5f);
  outp[512 + t] = finv;
  outp[768 + t] = fb[t] - fm[t] * finv;
  if (t < 64) {
    float ginv = gs[t] * rsqrtf(gv[t] + 1e-5f);
    outp[1024 + t] = ginv;
    outp[1088 + t] = gb[t] - gm[t] * ginv;
  }
}

// ---------------------------------------------------------------------------
// Generic WMMA GEMM over channels of an NCHW tensor.
//   out[b,o,sp] = epi( sum_c W[o,c] * act(in[b,c,sp]) )
//   act: optional (in0+in1) [compile-time], optional BN fold A*x+B [compile-time]
//   epi: +bias -> optional GELU -> optional *mul[b,o,sp] -> optional *ls[o]
//        -> optional +res[b,o,sp]   (uniform runtime branches, once per tile)
// One wave computes a 16(o) x 16(sp) tile with v_wmma_f32_16x16x32_bf16.
// Block = 256 threads = 8 waves; grid = (M/16, Cout/128).
// ---------------------------------------------------------------------------
template <bool HAS_IN1, bool HAS_BN>
__global__ void gemm_wmma_kernel(const float* __restrict__ in0, const float* __restrict__ in1,
                                 const float* __restrict__ bnA, const float* __restrict__ bnB,
                                 const float* __restrict__ W, int ldw, int Cin, int Cout,
                                 const float* __restrict__ bias, int act_gelu,
                                 const float* __restrict__ mul, const float* __restrict__ ls,
                                 const float* res, float* out) {
  int tid  = threadIdx.x;
  int wave = tid >> 5;
  int lane = tid & 31;
  int ln   = lane & 15;
  int hiA  = (lane & 16) ? 8 : 0;    // 32-bit D layout: lanes 16-31 hold rows M+8
  int hiB  = (lane & 16) ? 16 : 0;   // k order used here for A and B: lanes 16-31 hold K+16
  int o0 = (blockIdx.y * 8 + wave) * 16;
  int m0 = blockIdx.x * 16;
  int b  = m0 >> 12;                 // 4096 spatial positions per image
  int sp = m0 & 4095;
  const float* ib0 = in0 + (size_t)b * Cin * 4096 + sp + ln;
  const float* ib1 = HAS_IN1 ? in1 + (size_t)b * Cin * 4096 + sp + ln : nullptr;
  v8f acc = {};
  for (int k0 = 0; k0 < Cin; k0 += 32) {
    // A fragment: weights, row o = o0+ln, halves h -> K = hiB + h
    const float* wr = W + (size_t)(o0 + ln) * ldw + k0 + hiB;
    v16bf a;
#pragma unroll
    for (int h = 0; h < 8; h++)  a[h] = (__bf16)wr[h];
#pragma unroll
    for (int h = 8; h < 16; h++) a[h] = (__bf16)wr[h];   // wr[8..15]
    // B fragment: activations, col n = ln, halves h -> K = hiB + h  (branch-free)
    v16bf bm;
#pragma unroll
    for (int h = 0; h < 16; h++) {
      int c = k0 + hiB + h;
      float v = ib0[(size_t)c * 4096];
      if (HAS_IN1) v += ib1[(size_t)c * 4096];
      if (HAS_BN)  v = v * bnA[c] + bnB[c];
      bm[h] = (__bf16)v;
    }
    acc = __builtin_amdgcn_wmma_f32_16x16x32_bf16(false, a, false, bm, (short)0, acc,
                                                  false, false);
  }
  size_t obase = (size_t)b * Cout * 4096 + sp + ln;
#pragma unroll
  for (int r = 0; r < 8; r++) {
    int o = o0 + r + hiA;            // 32-bit D layout: VGPR r -> M=r (+8 for lanes 16-31)
    float v = acc[r];
    if (bias) v += bias[o];          // uniform (SGPR arg) branches
    if (act_gelu) v = gelu_exact(v);
    size_t oi = obase + (size_t)o * 4096;
    if (mul) v *= mul[oi];
    if (ls)  v *= ls[o];
    if (res) v += res[oi];
    out[oi] = v;
  }
}

// ---------------------------------------------------------------------------
// local_attn mixer input: T = dw3x3 + (1x5 then 5x1) + (1x11 then 11x1)
// One block handles a 16(h) x 64(w) strip of one (b,c) plane, LDS halo=5.
// ---------------------------------------------------------------------------
__global__ void local_mix_kernel(const float* y,
                                 const float* w33, const float* b33,
                                 const float* w01, const float* b01,
                                 const float* w02, const float* b02,
                                 const float* w11, const float* b11,
                                 const float* w12, const float* b12,
                                 float* T) {
  int bc = blockIdx.x;
  int c  = bc & 255;
  int h0 = blockIdx.y * 16;
  size_t base = (size_t)bc * 4096;
  __shared__ float tile[26 * 74];
  int tid = threadIdx.x;
  for (int i = tid; i < 26 * 74; i += 256) {
    int r = i / 74, cc = i % 74;
    int gh = h0 + r - 5, gw = cc - 5;
    float v = 0.f;
    if (gh >= 0 && gh < 64 && gw >= 0 && gw < 64) v = y[base + gh * 64 + gw];
    tile[i] = v;
  }
  __syncthreads();
  float k33[9], k01[5], k02[5], k11[11], k12[11];
#pragma unroll
  for (int i = 0; i < 9; i++)  k33[i] = w33[c * 9 + i];
#pragma unroll
  for (int i = 0; i < 5; i++)  { k01[i] = w01[c * 5 + i];  k02[i] = w02[c * 5 + i]; }
#pragma unroll
  for (int i = 0; i < 11; i++) { k11[i] = w11[c * 11 + i]; k12[i] = w12[c * 11 + i]; }
  float bb33 = b33[c], bb01 = b01[c], bb02 = b02[c], bb11 = b11[c], bb12 = b12[c];
  for (int p = 0; p < 4; p++) {
    int pix = tid + p * 256;
    int ph = pix >> 6, pw = pix & 63;
    int h = h0 + ph;
    float a33 = bb33;
#pragma unroll
    for (int i = 0; i < 3; i++)
#pragma unroll
      for (int j = 0; j < 3; j++)
        a33 += k33[i * 3 + j] * tile[(ph + i + 4) * 74 + pw + j + 4];
    float a15 = bb02;
#pragma unroll
    for (int i = 0; i < 5; i++) {
      int hh = h + i - 2;
      if (hh >= 0 && hh < 64) {       // second stage zero-pads the intermediate in h
        float inner = bb01;
#pragma unroll
        for (int j = 0; j < 5; j++) inner += k01[j] * tile[(ph + i + 3) * 74 + pw + j + 3];
        a15 += k02[i] * inner;
      }
    }
    float a111 = bb12;
#pragma unroll
    for (int i = 0; i < 11; i++) {
      int hh = h + i - 5;
      if (hh >= 0 && hh < 64) {
        float inner = bb11;
#pragma unroll
        for (int j = 0; j < 11; j++) inner += k11[j] * tile[(ph + i) * 74 + pw + j];
        a111 += k12[i] * inner;
      }
    }
    T[base + h * 64 + pw] = a33 + a15 + a111;
  }
}

// ---------------------------------------------------------------------------
// global_attn local_embed: grouped 4x4 stride-4 conv + BN + ReLU6 -> yg[B,64,16,16]
// ---------------------------------------------------------------------------
__global__ void ge_embed_kernel(const float* y, const float* gw, const float* gbias,
                                const float* gA, const float* gB, float* yg) {
  int idx = blockIdx.x * 256 + threadIdx.x;   // 16*64*256
  int b = idx >> 14;
  int g = (idx >> 8) & 63;
  int n = idx & 255;
  int i = n >> 4, j = n & 15;
  float s = gbias[g];
#pragma unroll
  for (int ic = 0; ic < 4; ic++) {
    const float* yp = y + ((size_t)(b * 256 + g * 4 + ic)) * 4096 + (i * 4) * 64 + j * 4;
    const float* wp = gw + (g * 4 + ic) * 16;
#pragma unroll
    for (int kh = 0; kh < 4; kh++)
#pragma unroll
      for (int kw = 0; kw < 4; kw++)
        s += wp[kh * 4 + kw] * yp[kh * 64 + kw];
  }
  s = s * gA[g] + gB[g];
  yg[idx] = fminf(fmaxf(s, 0.f), 6.f);
}

// qkv projection: [B,N,64] @ qkv_w^T -> q (scaled), k, v stored [B,HEADS,N,32]
__global__ void qkv_kernel(const float* yg, const float* qw, float* qb, float* kb, float* vb) {
  int idx = blockIdx.x * 256 + threadIdx.x;   // 16*256*192
  int b = idx / 49152;
  int r = idx - b * 49152;
  int n = r / 192;
  int o = r - n * 192;
  const float* yp = yg + (size_t)b * 16384 + n;
  float s = 0.f;
#pragma unroll
  for (int cc = 0; cc < 64; cc++) s += qw[o * 64 + cc] * yp[cc * 256];
  int kind = o >> 6, oo = o & 63, head = oo >> 5, d = oo & 31;
  size_t di = (((size_t)(b * 2 + head)) * 256 + n) * 32 + d;
  if (kind == 0)      qb[di] = s * 0.17677669529663687f;  // hd^-0.5
  else if (kind == 1) kb[di] = s;
  else                vb[di] = s;
}

// attention row: one block per (b,head,n); softmax over 256 keys + rel-pos bias
__global__ void attn_kernel(const float* qb, const float* kb, const float* vb,
                            const float* rpb, float* ao) {
  int bh = blockIdx.x >> 8;
  int n  = blockIdx.x & 255;
  int head = bh & 1;
  int b = bh >> 1;
  int m = threadIdx.x;
  const float* qp = qb + ((size_t)bh * 256 + n) * 32;
  const float* kp = kb + (size_t)bh * 256 * 32 + (size_t)m * 32;
  float lg = 0.f;
#pragma unroll
  for (int d = 0; d < 32; d++) lg += qp[d] * kp[d];
  int di = (n >> 4) - (m >> 4);
  int dj = (n & 15) - (m & 15);
  lg += rpb[((di + 15) * 31 + (dj + 15)) * 2 + head];
  __shared__ float red[256];
  __shared__ float pb[256];
  red[m] = lg; __syncthreads();
  for (int s = 128; s > 0; s >>= 1) { if (m < s) red[m] = fmaxf(red[m], red[m + s]); __syncthreads(); }
  float mx = red[0]; __syncthreads();
  float e = __expf(lg - mx);
  pb[m] = e; red[m] = e; __syncthreads();
  for (int s = 128; s > 0; s >>= 1) { if (m < s) red[m] += red[m + s]; __syncthreads(); }
  float inv = 1.0f / red[0];
  if (m < 32) {
    const float* vp = vb + (size_t)bh * 256 * 32 + m;
    float a = 0.f;
    for (int mm = 0; mm < 256; mm++) a += pb[mm] * vp[(size_t)mm * 32];
    ao[((size_t)b * 256 + n) * 64 + head * 32 + m] = a * inv;  // [b,n,c]
  }
}

// attention output projection -> og[B,64,16,16] (NCHW)
__global__ void gproj_kernel(const float* ao, const float* pw, const float* pbias, float* og) {
  int idx = blockIdx.x * 256 + threadIdx.x;   // 16*64*256
  int b = idx >> 14;
  int c = (idx >> 8) & 63;
  int n = idx & 255;
  const float* ap = ao + ((size_t)b * 256 + n) * 64;
  float s = pbias[c];
#pragma unroll
  for (int k = 0; k < 64; k++) s += pw[c * 64 + k] * ap[k];
  og[(size_t)(b * 64 + c) * 256 + n] = s;
}

// gout 1x1 (64->256) at 16x16 resolution -> gs[B,256,16,16]
__global__ void gout_kernel(const float* og, const float* w, const float* bias, float* gsb) {
  int idx = blockIdx.x * 256 + threadIdx.x;   // 16*256*256
  int b = idx >> 16;
  int o = (idx >> 8) & 255;
  int n = idx & 255;
  const float* op = og + (size_t)b * 16384 + n;
  float s = bias[o];
#pragma unroll
  for (int cc = 0; cc < 64; cc++) s += w[o * 64 + cc] * op[cc * 256];
  gsb[(size_t)(b * 256 + o) * 256 + n] = s;
}

// bilinear x4 upsample, align_corners=True: 16x16 -> 64x64
__global__ void up4_kernel(const float* gsrc, float* xg) {
  int idx = blockIdx.x * 256 + threadIdx.x;   // 16*256*4096
  int w = idx & 63, h = (idx >> 6) & 63;
  int bo = idx >> 12;
  float shf = h * (15.0f / 63.0f);
  int i0 = (int)shf; float f = shf - i0; int i1 = i0 + 1 > 15 ? 15 : i0 + 1;
  float swf = w * (15.0f / 63.0f);
  int j0 = (int)swf; float g = swf - j0; int j1 = j0 + 1 > 15 ? 15 : j0 + 1;
  const float* p = gsrc + (size_t)bo * 256;
  float v = (1.f - f) * ((1.f - g) * p[i0 * 16 + j0] + g * p[i0 * 16 + j1]) +
            f * ((1.f - g) * p[i1 * 16 + j0] + g * p[i1 * 16 + j1]);
  xg[idx] = v;
}

// FFN depthwise 3x3 + exact GELU (512-channel chunk)
__global__ void ffn_dw_kernel(const float* f1, const float* w, const float* bias, float* g) {
  int bc = blockIdx.x;            // b*512 + c
  int c  = bc & 511;
  int h0 = blockIdx.y * 16;
  size_t base = (size_t)bc * 4096;
  __shared__ float tile[18 * 66];
  int tid = threadIdx.x;
  for (int i = tid; i < 18 * 66; i += 256) {
    int r = i / 66, cc = i % 66;
    int gh = h0 + r - 1, gw = cc - 1;
    float v = 0.f;
    if (gh >= 0 && gh < 64 && gw >= 0 && gw < 64) v = f1[base + gh * 64 + gw];
    tile[i] = v;
  }
  __syncthreads();
  float k[9];
#pragma unroll
  for (int i = 0; i < 9; i++) k[i] = w[c * 9 + i];
  float bb = bias[c];
  for (int p = 0; p < 4; p++) {
    int pix = tid + p * 256;
    int ph = pix >> 6, pw = pix & 63;
    float a = bb;
#pragma unroll
    for (int i = 0; i < 3; i++)
#pragma unroll
      for (int j = 0; j < 3; j++)
        a += k[i * 3 + j] * tile[(ph + i) * 66 + pw + j];
    g[base + (h0 + ph) * 64 + pw] = gelu_exact(a);
  }
}

// ---------------------------------------------------------------------------
extern "C" void kernel_launch(void* const* d_in, const int* in_sizes, int n_in,
                              void* d_out, int out_size, void* d_ws, size_t ws_size,
                              hipStream_t stream) {
  const float* x       = (const float*)d_in[0];
  const float* bn1_s   = (const float*)d_in[1];
  const float* bn1_b   = (const float*)d_in[2];
  const float* bn1_m   = (const float*)d_in[3];
  const float* bn1_v   = (const float*)d_in[4];
  const float* proj1_w = (const float*)d_in[5];
  const float* proj1_b = (const float*)d_in[6];
  const float* la0_w   = (const float*)d_in[7];
  const float* la0_b   = (const float*)d_in[8];
  const float* la01_w  = (const float*)d_in[9];
  const float* la01_b  = (const float*)d_in[10];
  const float* la02_w  = (const float*)d_in[11];
  const float* la02_b  = (const float*)d_in[12];
  const float* la11_w  = (const float*)d_in[13];
  const float* la11_b  = (const float*)d_in[14];
  const float* la12_w  = (const float*)d_in[15];
  const float* la12_b  = (const float*)d_in[16];
  const float* la3_w   = (const float*)d_in[17];
  const float* la3_b   = (const float*)d_in[18];
  const float* ge_w    = (const float*)d_in[19];
  const float* ge_b    = (const float*)d_in[20];
  const float* gbn_s   = (const float*)d_in[21];
  const float* gbn_b   = (const float*)d_in[22];
  const float* gbn_m   = (const float*)d_in[23];
  const float* gbn_v   = (const float*)d_in[24];
  const float* qkv_w   = (const float*)d_in[25];
  const float* rpb     = (const float*)d_in[26];
  const float* gproj_w = (const float*)d_in[27];
  const float* gproj_b = (const float*)d_in[28];
  const float* gout_w  = (const float*)d_in[29];
  const float* gout_b  = (const float*)d_in[30];
  const float* proj2_w = (const float*)d_in[31];
  const float* proj2_b = (const float*)d_in[32];
  const float* ls1     = (const float*)d_in[33];
  const float* fbn_s   = (const float*)d_in[34];
  const float* fbn_b   = (const float*)d_in[35];
  const float* fbn_m   = (const float*)d_in[36];
  const float* fbn_v   = (const float*)d_in[37];
  const float* fc1_w   = (const float*)d_in[38];
  const float* fc1_b   = (const float*)d_in[39];
  const float* fdw_w   = (const float*)d_in[40];
  const float* fdw_b   = (const float*)d_in[41];
  const float* fc2_w   = (const float*)d_in[42];
  const float* fc2_b   = (const float*)d_in[43];
  const float* ls2     = (const float*)d_in[44];
  float* out = (float*)d_out;

  float* ws = (float*)d_ws;
  float* Y   = ws;                 // gelu(proj1(bn(x)))           67 MB
  float* T   = ws + NY;            // local mixer pre-1x1          67 MB
  float* XD  = ws + 2 * NY;        // local branch output          67 MB
  float* XG  = ws + 3 * NY;        // global branch output         67 MB
  float* SM  = ws + 4 * NY;        // small buffers
  float* bn1A = SM,          *bn1B = SM + 256;
  float* fbnA = SM + 512,    *fbnB = SM + 768;
  float* gA   = SM + 1024,   *gB   = SM + 1088;
  float* YG = SM + 2048;           // 16*64*256
  float* QB = YG + 262144;
  float* KB = QB + 262144;
  float* VB = KB + 262144;
  float* AO = VB + 262144;
  float* OG = AO + 262144;
  float* GS = OG + 262144;         // 16*256*256
  float* AT = GS + 1048576;        // attn_out residual tensor     67 MB
  float* F1 = ws;                  // FFN 512-ch chunk (reuses Y,T)
  float* G  = ws + 2 * NY;         // FFN post-dw chunk (reuses XD,XG)

  bnprep_kernel<<<1, 256, 0, stream>>>(bn1_s, bn1_b, bn1_m, bn1_v,
                                       fbn_s, fbn_b, fbn_m, fbn_v,
                                       gbn_s, gbn_b, gbn_m, gbn_v, SM);
  dim3 g256(4096, 2), g512(4096, 4);
  // y = gelu(proj1(bn1(x)))
  gemm_wmma_kernel<false, true><<<g256, 256, 0, stream>>>(
      x, nullptr, bn1A, bn1B, proj1_w, 256, 256, 256,
      proj1_b, 1, nullptr, nullptr, nullptr, Y);
  // T = x33 + x15 + x111
  local_mix_kernel<<<dim3(4096, 4), 256, 0, stream>>>(Y, la0_w, la0_b, la01_w, la01_b,
                                                      la02_w, la02_b, la11_w, la11_b,
                                                      la12_w, la12_b, T);
  // xd = (la3 @ T + b) * y
  gemm_wmma_kernel<false, false><<<g256, 256, 0, stream>>>(
      T, nullptr, nullptr, nullptr, la3_w, 256, 256, 256,
      la3_b, 0, Y, nullptr, nullptr, XD);
  // global branch
  ge_embed_kernel<<<1024, 256, 0, stream>>>(Y, ge_w, ge_b, gA, gB, YG);
  qkv_kernel<<<3072, 256, 0, stream>>>(YG, qkv_w, QB, KB, VB);
  attn_kernel<<<8192, 256, 0, stream>>>(QB, KB, VB, rpb, AO);
  gproj_kernel<<<1024, 256, 0, stream>>>(AO, gproj_w, gproj_b, OG);
  gout_kernel<<<4096, 256, 0, stream>>>(OG, gout_w, gout_b, GS);
  up4_kernel<<<65536, 256, 0, stream>>>(GS, XG);
  // attn_out = proj2(xd + xg) * ls1 + x
  gemm_wmma_kernel<true, false><<<g256, 256, 0, stream>>>(
      XD, XG, nullptr, nullptr, proj2_w, 256, 256, 256,
      proj2_b, 0, nullptr, ls1, x, AT);
  // FFN in two 512-channel chunks; fc2 accumulates into d_out
  for (int chunk = 0; chunk < 2; chunk++) {
    gemm_wmma_kernel<false, true><<<g512, 256, 0, stream>>>(
        AT, nullptr, fbnA, fbnB,
        fc1_w + (size_t)chunk * 512 * 256, 256, 256, 512,
        fc1_b + chunk * 512, 0, nullptr, nullptr, nullptr, F1);
    ffn_dw_kernel<<<dim3(8192, 4), 256, 0, stream>>>(F1, fdw_w + (size_t)chunk * 512 * 9,
                                                     fdw_b + chunk * 512, G);
    gemm_wmma_kernel<false, false><<<g256, 256, 0, stream>>>(
        G, nullptr, nullptr, nullptr,
        fc2_w + chunk * 512, 1024, 512, 256,
        chunk == 0 ? fc2_b : nullptr, 0, nullptr, ls2,
        chunk == 0 ? AT : out, out);
  }
}
